// MultiHeadAttention_34024730919067
// MI455X (gfx1250) — matrix operand order repaired and checked by reading, not compiled, not run on hardware
//
#include <hip/hip_runtime.h>

#define DM 2048   // d_model
#define NH 16     // heads
#define HD 128    // head dim
#define BB 2      // batch
#define TT 2048   // seq len
#define BT (BB*TT)

typedef __bf16 bf16;
typedef __attribute__((ext_vector_type(4)))  bf16  v4bf;
typedef __attribute__((ext_vector_type(8)))  bf16  v8bf;
typedef __attribute__((ext_vector_type(16))) bf16  v16bf;
typedef __attribute__((ext_vector_type(8)))  float v8f;

// Types matching the async-copy builtin's parameters:
//   (vector_size(16) int __device__*, vector_size(16) int __shared__*, imm, imm)
typedef int v4i_async __attribute__((vector_size(16)));
typedef __attribute__((address_space(1))) v4i_async* gv4i_p;
typedef __attribute__((address_space(3))) v4i_async* lv4i_p;

#if defined(__has_builtin)
#if __has_builtin(__builtin_amdgcn_global_load_async_to_lds_b128)
#define HAVE_ASYNC_B128 1
#endif
#if __has_builtin(__builtin_amdgcn_s_wait_asynccnt)
#define HAVE_WAIT_ASYNC 1
#endif
#endif

// Async DMA: 16 bytes per lane, global -> LDS, tracked with ASYNCcnt.
__device__ __forceinline__ void async_copy_b128(const bf16* g, bf16* l) {
  unsigned loff = (unsigned)(unsigned long long)l;   // LDS aperture low 32 bits = LDS offset
#if defined(HAVE_ASYNC_B128)
  __builtin_amdgcn_global_load_async_to_lds_b128(
      (gv4i_p)(unsigned long long)g, (lv4i_p)loff, 0, 0);
#else
  unsigned long long ga = (unsigned long long)g;
  asm volatile("global_load_async_to_lds_b128 %0, %1, off"
               :: "v"(loff), "v"(ga) : "memory");
#endif
}
__device__ __forceinline__ void wait_async0() {
#if defined(HAVE_WAIT_ASYNC)
  __builtin_amdgcn_s_wait_asynccnt(0);
#else
  asm volatile("s_wait_asynccnt 0" ::: "memory");
#endif
}

__device__ __forceinline__ v16bf join8(v8bf lo, v8bf hi) {
  v16bf r;
#pragma unroll
  for (int i = 0; i < 8; ++i) { r[i] = lo[i]; r[i + 8] = hi[i]; }
  return r;
}
// A-fragment (16x32 bf16, ISA 7.12.2): lane holds K = kb..kb+7 and kb+16..kb+23,
// kb = k0 + (lane>=16 ? 8 : 0). p points at element [m][kb].
__device__ __forceinline__ v16bf load_a(const bf16* p) {
  return join8(*(const v8bf*)p, *(const v8bf*)(p + 16));
}
// B-fragment (32x16): lane holds column N, K = kb..kb+7 and kb+16..kb+23,
// kb = k0 + (lane>=16 ? 8 : 0). p points at element [n][kb].
__device__ __forceinline__ v16bf load_b(const bf16* p) {
  return join8(*(const v8bf*)p, *(const v8bf*)(p + 16));
}
__device__ __forceinline__ v8f wmma_bf16(v16bf a, v16bf b, v8f c) {
  return __builtin_amdgcn_wmma_f32_16x16x32_bf16(false, a, false, b, (short)0, c,
                                                 false, false);
}
__device__ __forceinline__ v8f zero8() {
  v8f z;
#pragma unroll
  for (int i = 0; i < 8; ++i) z[i] = 0.f;
  return z;
}

// Stage a 128x32 bf16 tile (row stride DM in global) into LDS [128][32], 256 threads.
__device__ __forceinline__ void stage_tile(const bf16* gbase, bf16* lbase, int tid) {
#pragma unroll
  for (int it = 0; it < 2; ++it) {
    int c   = it * 256 + tid;     // 512 chunks of 8 elements (16B)
    int ml  = c >> 2;
    int kl0 = (c & 3) * 8;
    async_copy_b128(gbase + (size_t)ml * DM + kl0, lbase + ml * 32 + kl0);
  }
}

// ---------------- fp32 -> bf16 elementwise ----------------
__global__ void k_cvt(const float* __restrict__ src, bf16* __restrict__ dst, int n4) {
  int i = blockIdx.x * blockDim.x + threadIdx.x;
  if (i < n4) {
    float4 v = ((const float4*)src)[i];
    v4bf o;
    o[0] = (bf16)v.x; o[1] = (bf16)v.y; o[2] = (bf16)v.z; o[3] = (bf16)v.w;
    ((v4bf*)dst)[i] = o;
  }
}

// ---------------- fp32 [m][R][Ccols] -> bf16 transposed [m][Ccols][R] ----------------
__global__ void k_transpose_cvt(const float* __restrict__ src, bf16* __restrict__ dst,
                                int R, int Ccols) {
  int m = blockIdx.y;
  int idx = blockIdx.x * blockDim.x + threadIdx.x;
  if (idx < R * Ccols) {
    int i = idx % R;       // contiguous in dst
    int j = idx / R;
    dst[(size_t)(m * Ccols + j) * R + i] = (bf16)src[(size_t)(m * R + i) * Ccols + j];
  }
}

// ---------------- QKV projection: [BT,DM] x [DM,HD] per (which,h) ----------------
__global__ void __launch_bounds__(256)
k_qkv(const bf16* __restrict__ Xh, const bf16* __restrict__ Wt,
      const float* __restrict__ bq, const float* __restrict__ bk,
      const float* __restrict__ bv, bf16* __restrict__ QKV) {
  __shared__ __attribute__((aligned(32))) bf16 Xs[2][128 * 32];
  __shared__ __attribute__((aligned(32))) bf16 Ws[2][128 * 32];
  const int tid  = threadIdx.x;
  const int lane = tid & 31;
  const int wave = tid >> 5;
  const int cl = lane & 15;
  const int half = lane >> 4;
  const int wh = blockIdx.y;              // which*NH + h
  const int which = wh >> 4;
  const int h = wh & 15;
  const int r0 = blockIdx.x * 128;        // row block in [BT]

  const bf16* xbase = Xh + (size_t)r0 * DM;
  const bf16* wbase = Wt + (size_t)wh * HD * DM;   // [HD rows=N][DM cols=K]
  v8f acc[8];
#pragma unroll
  for (int t = 0; t < 8; ++t) acc[t] = zero8();

  stage_tile(xbase, Xs[0], tid);          // k0 = 0
  stage_tile(wbase, Ws[0], tid);

  const int NKS = DM / 32;                // 64
  for (int ks = 0; ks < NKS; ++ks) {
    const int buf = ks & 1;
    wait_async0();                        // my async copies for `buf` landed
    __syncthreads();                      // everyone's landed; prev buf fully read
    if (ks + 1 < NKS) {                   // prefetch next tile into other buffer
      stage_tile(xbase + (ks + 1) * 32, Xs[buf ^ 1], tid);
      stage_tile(wbase + (ks + 1) * 32, Ws[buf ^ 1], tid);
    }
    v16bf a = load_a(Xs[buf] + (wave * 16 + cl) * 32 + half * 8);
#pragma unroll
    for (int t = 0; t < 8; ++t) {
      const bf16* bp = Ws[buf] + (t * 16 + cl) * 32 + half * 8;
      acc[t] = wmma_bf16(a, load_b(bp), acc[t]);
    }
  }

  const float* bias = (which == 0) ? bq : ((which == 1) ? bk : bv);
  const int bId = r0 / TT;
  const int trow0 = (r0 + wave * 16) % TT;
  bf16* obase = QKV + ((size_t)(which * BB + bId) * NH + h) * TT * HD;
#pragma unroll
  for (int t = 0; t < 8; ++t) {
    float bvv = bias[h * HD + t * 16 + cl];
#pragma unroll
    for (int r = 0; r < 8; ++r) {
      int trow = trow0 + r + half * 8;   // C/D layout: VGPR r -> row r (+8 upper half)
      obase[(size_t)trow * HD + t * 16 + cl] = (bf16)(acc[t][r] + bvv);
    }
  }
}

// ---------------- Flash attention: one (b,h,128-row tile) per block ----------------
__global__ void __launch_bounds__(256)
k_flash(const bf16* __restrict__ QKV, bf16* __restrict__ Yc) {
  __shared__ __attribute__((aligned(32))) bf16 Ks[32 * HD];     // [s_local][d] row-major
  __shared__ __attribute__((aligned(32))) bf16 Vt[HD * 32];     // [d][s_local] transposed
  __shared__ __attribute__((aligned(32))) bf16 Pb[8 * 16 * 32]; // per-wave P 16x32
  const int tid  = threadIdx.x;
  const int lane = tid & 31;
  const int wave = tid >> 5;
  const int cl = lane & 15;
  const int half = lane >> 4;
  const int t0 = blockIdx.x * 128;
  const int h = blockIdx.y;
  const int b = blockIdx.z;
  const int m0 = t0 + wave * 16;
  const float scale = 0.08838834764831845f;  // 1/sqrt(128)

  const bf16* Qb = QKV + ((size_t)(0 * BB + b) * NH + h) * TT * HD;
  const bf16* Kb = QKV + ((size_t)(1 * BB + b) * NH + h) * TT * HD;
  const bf16* Vb = QKV + ((size_t)(2 * BB + b) * NH + h) * TT * HD;

  v16bf qf[4];
  {
    const bf16* qrow = Qb + (size_t)(m0 + cl) * HD + half * 8;
#pragma unroll
    for (int kc = 0; kc < 4; ++kc) qf[kc] = load_a(qrow + kc * 32);
  }
  v8f y[8];
#pragma unroll
  for (int t = 0; t < 8; ++t) y[t] = zero8();
  float mrow[8], lrow[8];
#pragma unroll
  for (int r = 0; r < 8; ++r) { mrow[r] = -1e30f; lrow[r] = 0.f; }

  bf16* pb = Pb + wave * (16 * 32);
  const int sEnd = t0 + 128;
  for (int s0 = 0; s0 < sEnd; s0 += 32) {
    __syncthreads();                       // previous Ks/Vt readers are done
    // K chunk (contiguous 32x128) -> LDS via async DMA
#pragma unroll
    for (int it = 0; it < 2; ++it) {
      int c = it * 256 + tid;              // 512 chunks of 8 elements
      async_copy_b128(Kb + (size_t)s0 * HD + c * 8, Ks + c * 8);
    }
    // V chunk -> LDS transposed (vectorized 16B reads, b16 scatter stores)
#pragma unroll
    for (int it = 0; it < 2; ++it) {
      int linear = it * 2048 + tid * 8;
      int sl = linear >> 7;                // 0..31
      int d0 = linear & 127;
      v8bf v = *(const v8bf*)(Vb + (size_t)(s0 + sl) * HD + d0);
#pragma unroll
      for (int j = 0; j < 8; ++j) Vt[(d0 + j) * 32 + sl] = v[j];
    }
    if (s0 + 32 < sEnd) __builtin_prefetch(Vb + (size_t)(s0 + 32) * HD, 0, 3);
    wait_async0();                         // my K copies are in LDS
    __syncthreads();                       // everyone's K/V staging visible

    // S = Q K^T for two 16x16 tiles, K fragments from LDS (row s, contiguous d)
    float st[2][8];
#pragma unroll
    for (int nt = 0; nt < 2; ++nt) {
      v8f sacc = zero8();
#pragma unroll
      for (int kc = 0; kc < 4; ++kc) {
        const bf16* kp = Ks + (nt * 16 + cl) * HD + kc * 32 + half * 8;
        sacc = wmma_bf16(qf[kc], load_b(kp), sacc);
      }
      int col = s0 + nt * 16 + cl;
#pragma unroll
      for (int r = 0; r < 8; ++r) {
        int row = m0 + r + half * 8;
        st[nt][r] = (col <= row) ? sacc[r] * scale : -1e30f;
      }
    }
    // online softmax stats (rows striped across 16 lanes of each half-wave)
    float al[8];
#pragma unroll
    for (int r = 0; r < 8; ++r) {
      float v = fmaxf(st[0][r], st[1][r]);
#pragma unroll
      for (int off = 1; off < 16; off <<= 1) v = fmaxf(v, __shfl_xor(v, off, 32));
      float mn = fmaxf(mrow[r], v);
      al[r] = __expf(mrow[r] - mn);
      mrow[r] = mn;
      st[0][r] = __expf(st[0][r] - mn);
      st[1][r] = __expf(st[1][r] - mn);
      float ss = st[0][r] + st[1][r];
#pragma unroll
      for (int off = 1; off < 16; off <<= 1) ss += __shfl_xor(ss, off, 32);
      lrow[r] = lrow[r] * al[r] + ss;
    }
#pragma unroll
    for (int t = 0; t < 8; ++t)
#pragma unroll
      for (int r = 0; r < 8; ++r) y[t][r] *= al[r];

    // P (C/D layout) -> wave-private LDS -> bf16 A-fragment
#pragma unroll
    for (int r = 0; r < 8; ++r) {
      pb[(r + half * 8) * 32 + cl]      = (bf16)st[0][r];
      pb[(r + half * 8) * 32 + 16 + cl] = (bf16)st[1][r];
    }
    asm volatile("s_wait_dscnt 0" ::: "memory");
    v16bf pf = load_a(pb + cl * 32 + half * 8);
#pragma unroll
    for (int t = 0; t < 8; ++t) {          // Y += P @ V  (V^T staged in LDS)
      const bf16* vp = Vt + (t * 16 + cl) * 32 + half * 8;
      y[t] = wmma_bf16(pf, load_b(vp), y[t]);
    }
  }

  bf16* ob = Yc + (size_t)b * TT * DM + h * HD;
#pragma unroll
  for (int r = 0; r < 8; ++r) {
    float inv = 1.f / lrow[r];
    int trow = m0 + r + half * 8;
#pragma unroll
    for (int t = 0; t < 8; ++t)
      ob[(size_t)trow * DM + t * 16 + cl] = (bf16)(y[t][r] * inv);
  }
}

// ---------------- Output projection: [BT,DM] x Wo[DM,DM] + bo -> fp32 ----------------
__global__ void __launch_bounds__(256)
k_oproj(const bf16* __restrict__ Yc, const bf16* __restrict__ Wot,
        const float* __restrict__ bo, float* __restrict__ out) {
  __shared__ __attribute__((aligned(32))) bf16 Xs[2][128 * 32];
  __shared__ __attribute__((aligned(32))) bf16 Ws[2][128 * 32];
  const int tid  = threadIdx.x;
  const int lane = tid & 31;
  const int wave = tid >> 5;
  const int cl = lane & 15;
  const int half = lane >> 4;
  const int r0 = blockIdx.x * 128;
  const int j0 = blockIdx.y * 128;

  const bf16* xbase = Yc + (size_t)r0 * DM;
  const bf16* wbase = Wot + (size_t)j0 * DM;
  v8f acc[8];
#pragma unroll
  for (int t = 0; t < 8; ++t) acc[t] = zero8();

  stage_tile(xbase, Xs[0], tid);
  stage_tile(wbase, Ws[0], tid);

  const int NKS = DM / 32;
  for (int ks = 0; ks < NKS; ++ks) {
    const int buf = ks & 1;
    wait_async0();
    __syncthreads();
    if (ks + 1 < NKS) {
      stage_tile(xbase + (ks + 1) * 32, Xs[buf ^ 1], tid);
      stage_tile(wbase + (ks + 1) * 32, Ws[buf ^ 1], tid);
    }
    v16bf a = load_a(Xs[buf] + (wave * 16 + cl) * 32 + half * 8);
#pragma unroll
    for (int t = 0; t < 8; ++t) {
      const bf16* bp = Ws[buf] + (t * 16 + cl) * 32 + half * 8;
      acc[t] = wmma_bf16(a, load_b(bp), acc[t]);
    }
  }
#pragma unroll
  for (int t = 0; t < 8; ++t) {
    float bb = bo[j0 + t * 16 + cl];
#pragma unroll
    for (int r = 0; r < 8; ++r) {
      int row = r0 + wave * 16 + r + half * 8;
      out[(size_t)row * DM + j0 + t * 16 + cl] = acc[t][r] + bb;
    }
  }
}

extern "C" void kernel_launch(void* const* d_in, const int* in_sizes, int n_in,
                              void* d_out, int out_size, void* d_ws, size_t ws_size,
                              hipStream_t stream) {
  (void)in_sizes; (void)n_in; (void)out_size; (void)ws_size;
  const float* x  = (const float*)d_in[0];
  const float* Wq = (const float*)d_in[1];
  const float* bq = (const float*)d_in[2];
  const float* Wk = (const float*)d_in[3];
  const float* bk = (const float*)d_in[4];
  const float* Wv = (const float*)d_in[5];
  const float* bv = (const float*)d_in[6];
  const float* Wo = (const float*)d_in[7];
  const float* bo = (const float*)d_in[8];
  float* out = (float*)d_out;

  char* ws = (char*)d_ws;
  size_t off = 0;
  bf16* Xh  = (bf16*)(ws + off); off += (size_t)BT * DM * 2;               // 16.8 MB
  bf16* Wt  = (bf16*)(ws + off); off += (size_t)3 * NH * HD * DM * 2;      // 25.2 MB
  bf16* Wot = (bf16*)(ws + off); off += (size_t)DM * DM * 2;               //  8.4 MB
  bf16* QKV = (bf16*)(ws + off); off += (size_t)3 * BB * NH * TT * HD * 2; // 50.3 MB
  bf16* Yc  = (bf16*)(ws + off); off += (size_t)BT * DM * 2;               // 16.8 MB

  k_cvt<<<(BT * DM / 4 + 255) / 256, 256, 0, stream>>>(x, Xh, BT * DM / 4);
  k_transpose_cvt<<<dim3((DM * HD + 255) / 256, NH), 256, 0, stream>>>(Wq, Wt, DM, HD);
  k_transpose_cvt<<<dim3((DM * HD + 255) / 256, NH), 256, 0, stream>>>(
      Wk, Wt + (size_t)NH * HD * DM, DM, HD);
  k_transpose_cvt<<<dim3((DM * HD + 255) / 256, NH), 256, 0, stream>>>(
      Wv, Wt + (size_t)2 * NH * HD * DM, DM, HD);
  k_transpose_cvt<<<dim3((DM * DM + 255) / 256, 1), 256, 0, stream>>>(Wo, Wot, DM, DM);

  k_qkv<<<dim3(BT / 128, 3 * NH), 256, 0, stream>>>(Xh, Wt, bq, bk, bv, QKV);
  k_flash<<<dim3(TT / 128, NH, BB), 256, 0, stream>>>(QKV, Yc);
  k_oproj<<<dim3(BT / 128, DM / 128), 256, 0, stream>>>(Yc, Wot, bo, out);
}
